// DS_Block_55637006352780
// MI455X (gfx1250) — matrix operand order repaired and checked by reading, not compiled, not run on hardware
//
#include <hip/hip_runtime.h>

#define BDIM 8
#define NPTS 2000
#define CCH 128
#define PPOOL 256
#define KNN_K 16
#define BN_SC 0.99999500003749969f /* 1/sqrt(1+1e-5) */

typedef __attribute__((ext_vector_type(16))) __bf16 v16bf;
typedef __attribute__((ext_vector_type(8)))  float  v8f;
typedef __attribute__((ext_vector_type(4)))  float  v4f;
typedef __attribute__((ext_vector_type(4)))  unsigned v4u;
typedef __attribute__((ext_vector_type(8)))  int    v8i;
typedef __attribute__((ext_vector_type(4)))  int    v4i;

__device__ __forceinline__ __bf16 f2bf(float f) { return (__bf16)f; }

__device__ __forceinline__ v8f wmma_bf(v16bf a, v16bf b, v8f c) {
  return __builtin_amdgcn_wmma_f32_16x16x32_bf16(false, a, false, b, (short)0, c,
                                                 false, false);
}

__device__ __forceinline__ float sigm(float x) { return 1.0f / (1.0f + __expf(-x)); }

// ---------------------------------------------------------------------------
// GEMM rr: Y[b][m][n] = act(scale*(sum_k A[b][m][k] * Bt[b][n][k] + bias[n]))
// Both operands row-major over k. Fast path: branch-free, float4 loads.
// Out-of-range rows are clamped (their outputs are never stored).
// ---------------------------------------------------------------------------
__global__ void __launch_bounds__(32) k_gemm_rr(
    const float* __restrict__ A, long sA, int lda,
    const float* __restrict__ Bt, long sB, int ldb,
    float* __restrict__ Y, long sY, int ldy,
    const float* __restrict__ bias,
    int M, int Nc, int K, float scale, int act)
{
  const int lane = threadIdx.x;
  const int b = blockIdx.z;
  const int m0 = blockIdx.y << 4, n0 = blockIdx.x << 4;
  const int half = lane >> 4, l15 = lane & 15;
  const int am = m0 + l15, bn = n0 + l15;
  const int gA = half << 3, gB = half << 3;
  const int amc = (am < M) ? am : (M - 1);
  const int bnc = (bn < Nc) ? bn : (Nc - 1);
  const float* Arow = A + sA * b + (long)amc * lda;
  const float* Brow = Bt + sB * b + (long)bnc * ldb;
  v8f acc;
#pragma unroll
  for (int i = 0; i < 8; ++i) acc[i] = 0.0f;
  const int K32 = K & ~31;
  int k0 = 0;
  for (; k0 < K32; k0 += 32) {
    v4f a0 = *(const v4f*)(Arow + k0 + gA);
    v4f a1 = *(const v4f*)(Arow + k0 + gA + 4);
    v4f a2 = *(const v4f*)(Arow + k0 + gA + 16);
    v4f a3 = *(const v4f*)(Arow + k0 + gA + 20);
    v4f b0 = *(const v4f*)(Brow + k0 + gB);
    v4f b1 = *(const v4f*)(Brow + k0 + gB + 4);
    v4f b2 = *(const v4f*)(Brow + k0 + gB + 16);
    v4f b3 = *(const v4f*)(Brow + k0 + gB + 20);
    v16bf a, bm;
#pragma unroll
    for (int j = 0; j < 4; ++j) {
      a[j]      = f2bf(a0[j]);
      a[j + 4]  = f2bf(a1[j]);
      a[j + 8]  = f2bf(a2[j]);
      a[j + 12] = f2bf(a3[j]);
      bm[j]      = f2bf(b0[j]);
      bm[j + 4]  = f2bf(b1[j]);
      bm[j + 8]  = f2bf(b2[j]);
      bm[j + 12] = f2bf(b3[j]);
    }
    acc = wmma_bf(a, bm, acc);
  }
  if (k0 < K) { // single tail iteration (K % 32 != 0): value selects, no branches
    v16bf a, bm;
#pragma unroll
    for (int j = 0; j < 16; ++j) {
      int ka = k0 + gA + j + ((j >= 8) ? 8 : 0);
      float av = Arow[(ka < K) ? ka : (K - 1)];
      a[j] = f2bf((ka < K) ? av : 0.0f);
      int kb = k0 + gB + j + ((j >= 8) ? 8 : 0);
      float bv = Brow[(kb < K) ? kb : (K - 1)];
      bm[j] = f2bf((kb < K) ? bv : 0.0f);
    }
    acc = wmma_bf(a, bm, acc);
  }
  float* Yb = Y + sY * b;
  const int rb = m0 + (half << 3);
  if (bn < Nc) {
#pragma unroll
    for (int i = 0; i < 8; ++i) {
      int m = rb + i;
      if (m < M) {
        float v = acc[i] + (bias ? bias[bn] : 0.0f);
        v *= scale;
        if (act == 1) v = fmaxf(v, 0.0f);
        else if (act == 2) v = (v > 20.0f) ? v : log1pf(__expf(v));
        Yb[(long)m * ldy + bn] = v;
      }
    }
  }
}

// ---------------------------------------------------------------------------
// GEMM cn: Y[b][m][n] = act(scale*(sum_k A[b][m][k] * X[b][k][n] + bias[m]))
// A row-major [M,K] (weights), X k-major (ldx). Used only for small CSM convs.
// ---------------------------------------------------------------------------
__global__ void __launch_bounds__(32) k_gemm_cn(
    const float* __restrict__ A, long sA, int lda,
    const float* __restrict__ X, long sX, int ldx,
    float* __restrict__ Y, long sY, int ldy,
    const float* __restrict__ bias,
    int M, int Nc, int K, float scale, int act)
{
  const int lane = threadIdx.x;
  const int b = blockIdx.z;
  const int m0 = blockIdx.y << 4, n0 = blockIdx.x << 4;
  const int half = lane >> 4, l15 = lane & 15;
  const int am = m0 + l15, bn = n0 + l15;
  const int gA = half << 3, gB = half << 3;
  const int amc = (am < M) ? am : (M - 1);
  const int bnc = (bn < Nc) ? bn : (Nc - 1);
  const float* Arow = A + sA * b + (long)amc * lda;
  const float* Xb = X + sX * b;
  v8f acc;
#pragma unroll
  for (int i = 0; i < 8; ++i) acc[i] = 0.0f;
  const int K32 = K & ~31;
  int k0 = 0;
  for (; k0 < K32; k0 += 32) {
    v4f a0 = *(const v4f*)(Arow + k0 + gA);
    v4f a1 = *(const v4f*)(Arow + k0 + gA + 4);
    v4f a2 = *(const v4f*)(Arow + k0 + gA + 16);
    v4f a3 = *(const v4f*)(Arow + k0 + gA + 20);
    v16bf a, bm;
#pragma unroll
    for (int j = 0; j < 4; ++j) {
      a[j]      = f2bf(a0[j]);
      a[j + 4]  = f2bf(a1[j]);
      a[j + 8]  = f2bf(a2[j]);
      a[j + 12] = f2bf(a3[j]);
    }
#pragma unroll
    for (int j = 0; j < 16; ++j) {
      int kb = k0 + gB + j + ((j >= 8) ? 8 : 0);
      bm[j] = f2bf(Xb[(long)kb * ldx + bnc]);
    }
    acc = wmma_bf(a, bm, acc);
  }
  if (k0 < K) {
    v16bf a, bm;
#pragma unroll
    for (int j = 0; j < 16; ++j) {
      int ka = k0 + gA + j + ((j >= 8) ? 8 : 0);
      float av = Arow[(ka < K) ? ka : (K - 1)];
      a[j] = f2bf((ka < K) ? av : 0.0f);
      int kb = k0 + gB + j + ((j >= 8) ? 8 : 0);
      float bv = Xb[(long)((kb < K) ? kb : (K - 1)) * ldx + bnc];
      bm[j] = f2bf((kb < K) ? bv : 0.0f);
    }
    acc = wmma_bf(a, bm, acc);
  }
  float* Yb = Y + sY * b;
  const int rb = m0 + (half << 3);
  if (bn < Nc) {
#pragma unroll
    for (int i = 0; i < 8; ++i) {
      int m = rb + i;
      if (m < M) {
        float v = acc[i] + (bias ? bias[m] : 0.0f);
        v *= scale;
        if (act == 1) v = fmaxf(v, 0.0f);
        else if (act == 2) v = (v > 20.0f) ? v : log1pf(__expf(v));
        Yb[(long)m * ldy + bn] = v;
      }
    }
  }
}

// ----------------------------- reductions / norms ---------------------------
__global__ void k_rowstats(const float* __restrict__ x, int S, float eps,
                           float* __restrict__ mean, float* __restrict__ rstd)
{
  int r = blockIdx.x;
  const float* row = x + (long)r * S;
  __shared__ float s1[256], s2[256];
  float a = 0.0f, b = 0.0f;
  for (int i = threadIdx.x; i < S; i += 256) { float v = row[i]; a += v; b += v * v; }
  s1[threadIdx.x] = a; s2[threadIdx.x] = b;
  __syncthreads();
  for (int o = 128; o > 0; o >>= 1) {
    if (threadIdx.x < o) { s1[threadIdx.x] += s1[threadIdx.x + o]; s2[threadIdx.x] += s2[threadIdx.x + o]; }
    __syncthreads();
  }
  if (threadIdx.x == 0) {
    float m = s1[0] / (float)S;
    float v = fmaxf(s2[0] / (float)S - m * m, 0.0f);
    mean[r] = m; rstd[r] = rsqrtf(v + eps);
  }
}

// stats over n for each (b,c); x is [B,N,C] -> coalesced column reduction
__global__ void k_colstats(const float* __restrict__ x, int N, int C, float eps,
                           float* __restrict__ mean, float* __restrict__ rstd, int total)
{
  int i = blockIdx.x * 256 + threadIdx.x;
  if (i >= total) return;
  int b = i / C, c = i % C;
  const float* p = x + (long)b * N * C + c;
  float s = 0.0f, s2 = 0.0f;
  for (int n = 0; n < N; ++n) { float v = p[(long)n * C]; s += v; s2 += v * v; }
  float m = s / (float)N;
  float var = fmaxf(s2 / (float)N - m * m, 0.0f);
  mean[i] = m; rstd[i] = rsqrtf(var + eps);
}

__global__ void k_apply_norm(const float* __restrict__ x, const float* __restrict__ mean,
                             const float* __restrict__ rstd, const float* __restrict__ res,
                             float* __restrict__ y, long total, int S, float scale, int act)
{
  long i = (long)blockIdx.x * 256 + threadIdx.x;
  if (i >= total) return;
  long r = i / S;
  float v = (x[i] - mean[r]) * rstd[r] * scale + (res ? res[i] : 0.0f);
  if (act) v = fmaxf(v, 0.0f);
  y[i] = v;
}

// instance norm apply on [B,N,C]; stats indexed by (b, c)
__global__ void k_apply_cnorm(const float* __restrict__ x, const float* __restrict__ mean,
                              const float* __restrict__ rstd, const float* __restrict__ res,
                              float* __restrict__ y, long total, int N, int C,
                              float scale, int act)
{
  long i = (long)blockIdx.x * 256 + threadIdx.x;
  if (i >= total) return;
  int c = (int)(i % C); long t = i / C; int b = (int)(t / N);
  int r = b * C + c;
  float v = (x[i] - mean[r]) * rstd[r] * scale + (res ? res[i] : 0.0f);
  if (act) v = fmaxf(v, 0.0f);
  y[i] = v;
}

__global__ void k_apply_ln(const float* __restrict__ x, const float* __restrict__ mean,
                           const float* __restrict__ rstd, const float* __restrict__ g,
                           const float* __restrict__ bt, float* __restrict__ y,
                           long total, int S)
{
  long i = (long)blockIdx.x * 256 + threadIdx.x;
  if (i >= total) return;
  long r = i / S; int c = (int)(i - r * S);
  y[i] = (x[i] - mean[r]) * rstd[r] * g[c] + bt[c];
}

// ----------------------------- elementwise ---------------------------------
__global__ void k_add(const float* __restrict__ a, const float* __restrict__ b,
                      float* __restrict__ y, long total)
{
  long i = (long)blockIdx.x * 256 + threadIdx.x;
  if (i < total) y[i] = a[i] + b[i];
}

__global__ void k_transpose(const float* __restrict__ in, float* __restrict__ out,
                            int Bn, int I, int J)
{
  long i = (long)blockIdx.x * 256 + threadIdx.x;
  long total = (long)Bn * I * J;
  if (i >= total) return;
  int j = (int)(i % J); long t = i / J; int ii = (int)(t % I); int b = (int)(t / I);
  out[((long)b * J + j) * I + ii] = in[i];
}

// [B,N,C]: reverse along N
__global__ void k_fliprows(const float* __restrict__ in, float* __restrict__ out,
                           int Bn, int N, int C)
{
  long i = (long)blockIdx.x * 256 + threadIdx.x;
  long total = (long)Bn * N * C;
  if (i >= total) return;
  int c = (int)(i % C); long t = i / C; int n = (int)(t % N); int b = (int)(t / N);
  out[((long)b * N + (N - 1 - n)) * C + c] = in[i];
}

// cat rows: [x_row | xu_row] -> rows of length 2C
__global__ void k_concat_rows(const float* __restrict__ x, const float* __restrict__ xu,
                              float* __restrict__ cat, long total, int C)
{
  long i = (long)blockIdx.x * 256 + threadIdx.x;
  if (i >= total) return;
  int c = (int)(i % (2 * C)); long t = i / (2 * C);
  cat[i] = (c < C) ? x[t * C + c] : xu[t * C + (c - C)];
}

__global__ void k_softmax_rows(float* __restrict__ x, int S)
{
  int r = blockIdx.x;
  float* row = x + (long)r * S;
  __shared__ float sm[256];
  float mx = -3.4e38f;
  for (int i = threadIdx.x; i < S; i += 256) mx = fmaxf(mx, row[i]);
  sm[threadIdx.x] = mx; __syncthreads();
  for (int o = 128; o > 0; o >>= 1) { if (threadIdx.x < o) sm[threadIdx.x] = fmaxf(sm[threadIdx.x], sm[threadIdx.x + o]); __syncthreads(); }
  mx = sm[0]; __syncthreads();
  float s = 0.0f;
  for (int i = threadIdx.x; i < S; i += 256) s += __expf(row[i] - mx);
  sm[threadIdx.x] = s; __syncthreads();
  for (int o = 128; o > 0; o >>= 1) { if (threadIdx.x < o) sm[threadIdx.x] += sm[threadIdx.x + o]; __syncthreads(); }
  float inv = 1.0f / sm[0];
  __syncthreads();
  for (int i = threadIdx.x; i < S; i += 256) row[i] = __expf(row[i] - mx) * inv;
}

// --------------------------------- Mamba -----------------------------------
__global__ void k_dwconv_silu(const float* __restrict__ xz, const float* __restrict__ w,
                              const float* __restrict__ bias, float* __restrict__ xc,
                              long total, int L, int E)
{
  long i = (long)blockIdx.x * 256 + threadIdx.x;
  if (i >= total) return;
  int e = (int)(i % E); long r = i / E; int l = (int)(r % L);
  float acc = bias[e];
#pragma unroll
  for (int k = 0; k < 4; ++k) {
    int ls = l - 3 + k;
    if (ls >= 0) acc += w[e * 4 + k] * xz[(r - 3 + k) * (long)(2 * E) + e];
  }
  xc[i] = acc * sigm(acc);
}

__global__ void k_scan(const float* __restrict__ dt, const float* __restrict__ xc,
                       const float* __restrict__ dbl, int ldd, int offB,
                       const float* __restrict__ A_log, float* __restrict__ ys,
                       int L, int E, int S)
{
  int b = blockIdx.x, c = threadIdx.x;
  __shared__ float sm[32]; // B then C (max 2*16)
  float h[16], a[16];
  for (int s = 0; s < S; ++s) { h[s] = 0.0f; a[s] = -__expf(A_log[(long)c * S + s]); }
  for (int t = 0; t < L; ++t) {
    long base = (long)b * L + t;
    if (c < 2 * S) sm[c] = dbl[base * ldd + offB + c];
    __syncthreads();
    float dtv = dt[base * E + c];
    float xv = xc[base * E + c];
    float acc = 0.0f;
    for (int s = 0; s < S; ++s) {
      float dA = __expf(dtv * a[s]);
      h[s] = dA * h[s] + dtv * xv * sm[s];
      acc += h[s] * sm[S + s];
    }
    ys[base * E + c] = acc;
    __syncthreads();
  }
}

__global__ void k_ygate(const float* __restrict__ ys, const float* __restrict__ xc,
                        const float* __restrict__ Dv, const float* __restrict__ xz,
                        float* __restrict__ y, long total, int E)
{
  long i = (long)blockIdx.x * 256 + threadIdx.x;
  if (i >= total) return;
  int e = (int)(i % E); long r = i / E;
  float v = ys[i] + xc[i] * Dv[e];
  float z = xz[r * (long)(2 * E) + E + e];
  y[i] = v * z * sigm(z);
}

// -------------------------------- stem / head -------------------------------
// output in [B,N,C] rows
__global__ void k_conv0(const float* __restrict__ x4, const float* __restrict__ W,
                        const float* __restrict__ bias, float* __restrict__ out,
                        long total, int C, int N)
{
  long i = (long)blockIdx.x * 256 + threadIdx.x;
  if (i >= total) return;
  int c = (int)(i % C); long t = i / C; int n = (int)(t % N); int b = (int)(t / N);
  const float* xp = x4 + ((long)b * N + n) * 4;
  float acc = bias[c];
#pragma unroll
  for (int j = 0; j < 4; ++j) acc += W[c * 4 + j] * xp[j];
  out[i] = fmaxf(acc * BN_SC, 0.0f);
}

__global__ void k_lin0(const float* __restrict__ x, const float* __restrict__ W,
                       const float* __restrict__ b0, float* __restrict__ out,
                       long total, int C)
{
  long i = (long)blockIdx.x * 256 + threadIdx.x;
  if (i >= total) return;
  const float* r = x + i * (long)C;
  float acc = b0[0];
  for (int c = 0; c < C; c += 4) {
    v4f v = *(const v4f*)(r + c);
    acc += W[c] * v[0] + W[c + 1] * v[1] + W[c + 2] * v[2] + W[c + 3] * v[3];
  }
  out[i] = acc;
}

// ----------------------------------- kNN -----------------------------------
// [B,N,C] rows; ranks by 2*dot - |y|^2 (same ordering as reference)
__global__ void k_knn(const float* __restrict__ x, int* __restrict__ idx, int C, int N)
{
  int n = blockIdx.x, b = blockIdx.y;
  __shared__ float q[CCH];
  __shared__ float red[256];
  __shared__ int redi[256];
  __shared__ int hp[256];
  __shared__ float alld[256][KNN_K];
  __shared__ int alli[256][KNN_K];
  const float* xb = x + (long)b * N * C;
  for (int c = threadIdx.x; c < C; c += 256) q[c] = xb[(long)n * C + c];
  hp[threadIdx.x] = 0;
  __syncthreads();
  float ld[KNN_K]; int li[KNN_K];
#pragma unroll
  for (int j = 0; j < KNN_K; ++j) { ld[j] = -3.4e38f; li[j] = 0; }
  for (int m = threadIdx.x; m < N; m += 256) {
    const float* r = xb + (long)m * C;
    float dot = 0.0f, nm = 0.0f;
    for (int c = 0; c < C; c += 4) {
      v4f v = *(const v4f*)(r + c);
      dot += q[c] * v[0] + q[c + 1] * v[1] + q[c + 2] * v[2] + q[c + 3] * v[3];
      nm += v[0] * v[0] + v[1] * v[1] + v[2] * v[2] + v[3] * v[3];
    }
    float d = 2.0f * dot - nm;
    if (d > ld[KNN_K - 1]) {
      int j = KNN_K - 1;
      while (j > 0 && ld[j - 1] < d) { ld[j] = ld[j - 1]; li[j] = li[j - 1]; --j; }
      ld[j] = d; li[j] = m;
    }
  }
#pragma unroll
  for (int j = 0; j < KNN_K; ++j) { alld[threadIdx.x][j] = ld[j]; alli[threadIdx.x][j] = li[j]; }
  __syncthreads();
  int* outp = idx + ((long)b * N + n) * KNN_K;
  for (int r = 0; r < KNN_K; ++r) {
    int p = hp[threadIdx.x];
    red[threadIdx.x] = (p < KNN_K) ? alld[threadIdx.x][p] : -3.4e38f;
    redi[threadIdx.x] = threadIdx.x;
    __syncthreads();
    for (int o = 128; o > 0; o >>= 1) {
      if (threadIdx.x < o && red[threadIdx.x + o] > red[threadIdx.x]) {
        red[threadIdx.x] = red[threadIdx.x + o]; redi[threadIdx.x] = redi[threadIdx.x + o];
      }
      __syncthreads();
    }
    int w = redi[0];
    if (threadIdx.x == w) { outp[r] = alli[threadIdx.x][hp[threadIdx.x]]; hp[threadIdx.x]++; }
    __syncthreads();
  }
}

// ------------------------------ fused DGCNN ---------------------------------
// [B,N,C] rows. One 8-wave block per point. The 16 kNN neighbor rows are
// gathered into LDS by the Tensor Data Mover (gather mode, 16-bit indices),
// then the edge tile feeds two WMMA GEMMs and a max over neighbors.
__global__ void __launch_bounds__(256) k_dgcnn(
    const float* __restrict__ x, const int* __restrict__ idx,
    const float* __restrict__ W1, const float* __restrict__ b1,
    const float* __restrict__ W2, const float* __restrict__ b2,
    float* __restrict__ out, int C, int N)
{
  int n = blockIdx.x, b = blockIdx.y;
  __shared__ float nb[KNN_K * CCH];      // TDM-gathered neighbor rows
  __shared__ float g[2 * CCH * KNN_K];   // [256 feat][16 nb]
  __shared__ float h1[CCH * KNN_K];
  __shared__ float h2[CCH * KNN_K];
  const float* xb = x + (long)b * N * C;
  const float* rowc = xb + (long)n * C;
  const int* ib = idx + ((long)b * N + n) * KNN_K;

  if (threadIdx.x < 32) { // wave 0 issues the TDM gather descriptor
    unsigned lds_off = (unsigned)(unsigned long long)(size_t)&nb[0];
    unsigned long long ga = (unsigned long long)(size_t)xb;
    v4u d0;
    d0[0] = 1u | (1u << 31);                         // count=1, gather_mode, 16b idx
    d0[1] = lds_off;                                 // LDS destination
    d0[2] = (unsigned)(ga & 0xffffffffull);          // global_addr[31:0]
    d0[3] = (unsigned)((ga >> 32) & 0x01ffffffull) | (2u << 30); // [56:32], type=2
    v8i d1;
    d1[0] = 0x20000;                                 // data_size = 4 bytes
    d1[1] = (int)((unsigned)C << 16);                // tensor_dim0 = C
    d1[2] = (int)(((unsigned)N & 0xffffu) << 16);    // tensor_dim1 = N
    d1[3] = (int)((unsigned)C << 16);                // tile_dim0 = C
    d1[4] = KNN_K;                                   // tile_dim1 = #indices
    d1[5] = C;                                       // tensor_dim0_stride = C
    d1[6] = 0; d1[7] = 0;
    v4i d2, d3;
#pragma unroll
    for (int k = 0; k < 4; ++k) {
      int i0 = __builtin_amdgcn_readfirstlane(ib[2 * k]);
      int i1 = __builtin_amdgcn_readfirstlane(ib[2 * k + 1]);
      d2[k] = (i0 & 0xffff) | (i1 << 16);
      int i2 = __builtin_amdgcn_readfirstlane(ib[8 + 2 * k]);
      int i3 = __builtin_amdgcn_readfirstlane(ib[8 + 2 * k + 1]);
      d3[k] = (i2 & 0xffff) | (i3 << 16);
    }
#if defined(__clang_major__) && (__clang_major__ >= 23)
    v8i dz; for (int k = 0; k < 8; ++k) dz[k] = 0;
    __builtin_amdgcn_tensor_load_to_lds(d0, d1, d2, d3, dz, 0);
#else
    __builtin_amdgcn_tensor_load_to_lds(d0, d1, d2, d3, 0);
#endif
    __builtin_amdgcn_s_wait_tensorcnt(0);
  }
  __syncthreads();
  {
    int t = threadIdx.x;
    if (t < C) {
      float ctr = rowc[t];
      for (int k = 0; k < KNN_K; ++k) g[t * KNN_K + k] = ctr;
    } else {
      int cc = t - C;
      float ctr = rowc[cc];
      for (int k = 0; k < KNN_K; ++k)
        g[t * KNN_K + k] = ctr - nb[k * C + cc];
    }
  }
  __syncthreads();
  const int wave = threadIdx.x >> 5, lane = threadIdx.x & 31;
  const int half = lane >> 4, l15 = lane & 15;
  const int gA = half << 3, gB = half << 3;
  // GEMM1: W1[128x256] x g[256x16] -> relu(BN*(.+b1)) -> h1
  {
    int m0 = wave << 4;
    const float* Arow = W1 + (long)(m0 + l15) * (2 * CCH);
    v8f acc;
#pragma unroll
    for (int i = 0; i < 8; ++i) acc[i] = 0.0f;
    for (int k0 = 0; k0 < 2 * CCH; k0 += 32) {
      v4f a0 = *(const v4f*)(Arow + k0 + gA);
      v4f a1 = *(const v4f*)(Arow + k0 + gA + 4);
      v4f a2 = *(const v4f*)(Arow + k0 + gA + 16);
      v4f a3 = *(const v4f*)(Arow + k0 + gA + 20);
      v16bf a, bm;
#pragma unroll
      for (int j = 0; j < 4; ++j) {
        a[j] = f2bf(a0[j]); a[j + 4] = f2bf(a1[j]);
        a[j + 8] = f2bf(a2[j]); a[j + 12] = f2bf(a3[j]);
      }
#pragma unroll
      for (int j = 0; j < 16; ++j) bm[j] = f2bf(g[(k0 + gB + j + ((j >= 8) ? 8 : 0)) * KNN_K + l15]);
      acc = wmma_bf(a, bm, acc);
    }
    int rb = m0 + (half << 3);
#pragma unroll
    for (int i = 0; i < 8; ++i)
      h1[(rb + i) * KNN_K + l15] = fmaxf(BN_SC * (acc[i] + b1[rb + i]), 0.0f);
  }
  __syncthreads();
  // GEMM2: W2[128x128] x h1[128x16] -> relu(BN*(.+b2)) -> h2
  {
    int m0 = wave << 4;
    const float* Arow = W2 + (long)(m0 + l15) * CCH;
    v8f acc;
#pragma unroll
    for (int i = 0; i < 8; ++i) acc[i] = 0.0f;
    for (int k0 = 0; k0 < CCH; k0 += 32) {
      v4f a0 = *(const v4f*)(Arow + k0 + gA);
      v4f a1 = *(const v4f*)(Arow + k0 + gA + 4);
      v4f a2 = *(const v4f*)(Arow + k0 + gA + 16);
      v4f a3 = *(const v4f*)(Arow + k0 + gA + 20);
      v16bf a, bm;
#pragma unroll
      for (int j = 0; j < 4; ++j) {
        a[j] = f2bf(a0[j]); a[j + 4] = f2bf(a1[j]);
        a[j + 8] = f2bf(a2[j]); a[j + 12] = f2bf(a3[j]);
      }
#pragma unroll
      for (int j = 0; j < 16; ++j) bm[j] = f2bf(h1[(k0 + gB + j + ((j >= 8) ? 8 : 0)) * KNN_K + l15]);
      acc = wmma_bf(a, bm, acc);
    }
    int rb = m0 + (half << 3);
#pragma unroll
    for (int i = 0; i < 8; ++i)
      h2[(rb + i) * KNN_K + l15] = fmaxf(BN_SC * (acc[i] + b2[rb + i]), 0.0f);
  }
  __syncthreads();
  if (threadIdx.x < (unsigned)C) {
    float mx = h2[threadIdx.x * KNN_K];
    for (int k = 1; k < KNN_K; ++k) mx = fmaxf(mx, h2[threadIdx.x * KNN_K + k]);
    out[((long)b * N + n) * C + threadIdx.x] = mx;
  }
}

// ============================ host orchestration ============================

struct MambaP {
  const float *A_log, *D, *conv_b, *conv_w, *dt_b, *dt_w, *in_proj, *out_proj, *x_proj;
  int Dm, E, S, R;
};
struct ResP { const float *W1, *W2, *b1, *b2; };
struct CSMP { const float *c1W, *c1b, *c3W, *c3b, *ln_b, *ln_g; MambaP sam; };
struct CSSMP {
  CSMP csm;
  const float *mb_ln_b, *mb_ln_g; MambaP mbm;
  const float *poolW, *poolb, *shotW, *shotb, *unpoolW, *unpoolb;
};

static inline unsigned ewg(long n) { return (unsigned)((n + 255) / 256); }

extern "C" void kernel_launch(void* const* d_in, const int* in_sizes, int n_in,
                              void* d_out, int out_size, void* d_ws, size_t ws_size,
                              hipStream_t stream)
{
  (void)in_sizes; (void)out_size; (void)ws_size;
  if (n_in < 184) return;

  // ---- parse params (x first, then jax tree_flatten order: sorted dict keys)
  int pi = 0;
  auto nx = [&]() { return (const float*)d_in[pi++]; };
  const float* x_in = nx();
  auto parse_mamba = [&](int Dm, int S) {
    MambaP m;
    m.A_log = nx(); m.D = nx(); m.conv_b = nx(); m.conv_w = nx();
    m.dt_b = nx(); m.dt_w = nx(); m.in_proj = nx(); m.out_proj = nx(); m.x_proj = nx();
    m.Dm = Dm; m.E = 2 * Dm; m.S = S; m.R = (Dm + 15) / 16;
    return m;
  };
  auto parse_res = [&]() { ResP r; r.W1 = nx(); r.W2 = nx(); r.b1 = nx(); r.b2 = nx(); return r; };

  const float *cfbm_ln_b, *cfbm_ln_g; MambaP cfbm_m;
  cfbm_ln_b = nx(); cfbm_ln_g = nx(); cfbm_m = parse_mamba(CCH, 16);
  const float* conv0W = nx(); const float* conv0b = nx();
  CSSMP cs[3];
  for (int i = 0; i < 3; ++i) {
    CSSMP c;
    c.csm.c1W = nx(); c.csm.c1b = nx(); c.csm.c3W = nx(); c.csm.c3b = nx();
    c.csm.ln_b = nx(); c.csm.ln_g = nx(); c.csm.sam = parse_mamba(PPOOL, 8);
    c.mb_ln_b = nx(); c.mb_ln_g = nx(); c.mbm = parse_mamba(CCH, 8);
    c.poolW = nx(); c.poolb = nx(); c.shotW = nx(); c.shotb = nx();
    c.unpoolW = nx(); c.unpoolb = nx();
    cs[i] = c;
  }
  ResP dg = parse_res();
  ResP embed1 = parse_res();
  ResP gcc_a[4]; for (int i = 0; i < 4; ++i) gcc_a[i] = parse_res();
  ResP gcc_b[4]; for (int i = 0; i < 4; ++i) gcc_b[i] = parse_res();
  const float* lin0W = nx(); const float* lin0b = nx();
  ResP lpfe_a[4]; for (int i = 0; i < 4; ++i) lpfe_a[i] = parse_res();
  ResP lpfe_b[4]; for (int i = 0; i < 4; ++i) lpfe_b[i] = parse_res();

  // ---- workspace (bump allocator, 256B aligned). Activations are [B,N,C].
  const long ACT = (long)BDIM * NPTS * CCH;
  const long EMBS = (long)BDIM * NPTS * PPOOL;
  const long PSZ = (long)BDIM * CCH * PPOOL;
  const long MROWS = (long)BDIM * NPTS;
  size_t off = 0;
  auto alloc = [&](long nf) {
    float* p = (float*)((char*)d_ws + off);
    off += (((size_t)nf * 4 + 255) / 256) * 256;
    return p;
  };
  float* cur = alloc(ACT);
  float* t1 = alloc(ACT);
  float* t2 = alloc(ACT);
  float* a_h = alloc(ACT);
  float* a_mo = alloc(ACT);
  float* f1 = alloc(ACT);
  float* f2 = alloc(ACT);
  float* xu = alloc(ACT);
  float* x12 = alloc(ACT);
  float* a_dg = alloc(ACT);
  float* xcn = alloc(ACT);                 // [B,C,N] copy for diff-pool
  float* m_xz = alloc(MROWS * 512);
  float* m_xc = alloc(MROWS * 256);
  float* m_dbl = alloc(MROWS * 48);
  float* m_dt = alloc(MROWS * 256);
  float* m_ys = alloc(MROWS * 256);
  float* m_yb = alloc(MROWS * 256);
  float* emb = alloc(EMBS);                // [B,P,N]
  float* emb2 = alloc(EMBS);               // [B,N,P] (also scratch for emb_r)
  float* catb = alloc((long)BDIM * NPTS * 2 * CCH);
  float* xd = alloc(PSZ);
  float* x2b = alloc(PSZ);
  float* c_pre = alloc(PSZ);
  float* c_o1 = alloc(PSZ);
  float* c_on = alloc(PSZ);
  float* c_mo = alloc(PSZ);
  float* stm = alloc(16384);
  float* str = alloc(16384);
  int* knn_idx = (int*)alloc((long)BDIM * NPTS * KNN_K);

  // ---- launch helpers
  auto gemm_rr = [&](const float* A, long sA, int lda, const float* Bt, long sB, int ldb,
                     float* Y, long sY, int ldy, const float* bias,
                     int M, int Nc, int K, float scale, int act, int batch) {
    dim3 g((Nc + 15) / 16, (M + 15) / 16, batch);
    k_gemm_rr<<<g, 32, 0, stream>>>(A, sA, lda, Bt, sB, ldb, Y, sY, ldy, bias, M, Nc, K, scale, act);
  };
  auto gemm_cn = [&](const float* A, long sA, int lda, const float* X, long sX, int ldx,
                     float* Y, long sY, int ldy, const float* bias,
                     int M, int Nc, int K, float scale, int act, int batch) {
    dim3 g((Nc + 15) / 16, (M + 15) / 16, batch);
    k_gemm_cn<<<g, 32, 0, stream>>>(A, sA, lda, X, sX, ldx, Y, sY, ldy, bias, M, Nc, K, scale, act);
  };
  auto rowstats = [&](const float* x, int R, int S, float eps) {
    k_rowstats<<<R, 256, 0, stream>>>(x, S, eps, stm, str);
  };
  auto colstats = [&](const float* x, float eps) {
    k_colstats<<<ewg(BDIM * CCH), 256, 0, stream>>>(x, NPTS, CCH, eps, stm, str, BDIM * CCH);
  };
  auto apply_norm = [&](const float* src, const float* res, float* dst, long R, int S,
                        float scale, int act) {
    long tot = R * S;
    k_apply_norm<<<ewg(tot), 256, 0, stream>>>(src, stm, str, res, dst, tot, S, scale, act);
  };
  auto apply_cnorm = [&](const float* src, const float* res, float* dst, float scale, int act) {
    k_apply_cnorm<<<ewg(ACT), 256, 0, stream>>>(src, stm, str, res, dst, ACT, NPTS, CCH, scale, act);
  };
  auto apply_ln = [&](const float* src, const float* g, const float* b, float* dst,
                      long R, int S) {
    long tot = R * S;
    k_apply_ln<<<ewg(tot), 256, 0, stream>>>(src, stm, str, g, b, dst, tot, S);
  };
  auto addk = [&](const float* a, const float* b, float* y, long tot) {
    k_add<<<ewg(tot), 256, 0, stream>>>(a, b, y, tot);
  };

  auto run_mamba = [&](const MambaP& m, const float* xin, float* out, int Bn, int L) {
    const int Dm = m.Dm, E = m.E, R = m.R, S = m.S;
    const int rows = Bn * L;
    const int ldd = R + 2 * S;
    gemm_rr(xin, 0, Dm, m.in_proj, 0, Dm, m_xz, 0, 2 * E, nullptr, rows, 2 * E, Dm, 1.0f, 0, 1);
    long totE = (long)rows * E;
    k_dwconv_silu<<<ewg(totE), 256, 0, stream>>>(m_xz, m.conv_w, m.conv_b, m_xc, totE, L, E);
    gemm_rr(m_xc, 0, E, m.x_proj, 0, E, m_dbl, 0, ldd, nullptr, rows, ldd, E, 1.0f, 0, 1);
    gemm_rr(m_dbl, 0, ldd, m.dt_w, 0, R, m_dt, 0, E, m.dt_b, rows, E, R, 1.0f, 2, 1);
    k_scan<<<Bn, E, 0, stream>>>(m_dt, m_xc, m_dbl, ldd, R, m.A_log, m_ys, L, E, S);
    k_ygate<<<ewg(totE), 256, 0, stream>>>(m_ys, m_xc, m.D, m_xz, m_yb, totE, E);
    gemm_rr(m_yb, 0, E, m.out_proj, 0, E, out, 0, Dm, nullptr, rows, Dm, E, 1.0f, 0, 1);
  };

  // rows-native mamba block (no transposes needed in [B,N,C] layout)
  auto mamba_block = [&](const float* ln_g, const float* ln_b, const MambaP& m,
                         const float* in, float* out) {
    rowstats(in, BDIM * NPTS, CCH, 1e-5f);
    apply_ln(in, ln_g, ln_b, a_h, BDIM * NPTS, CCH);
    run_mamba(m, a_h, a_mo, BDIM, NPTS);
    addk(a_mo, in, a_mo, ACT);
    rowstats(a_mo, BDIM * NPTS, CCH, 1e-5f);
    apply_ln(a_mo, ln_g, ln_b, out, BDIM * NPTS, CCH);
  };

  auto resblock = [&](const ResP& r, float* xio) {
    gemm_rr(xio, 0, CCH, r.W1, 0, CCH, t1, 0, CCH, r.b1, BDIM * NPTS, CCH, CCH, 1.0f, 0, 1);
    colstats(t1, 1e-5f);
    apply_cnorm(t1, nullptr, t2, BN_SC, 1);
    gemm_rr(t2, 0, CCH, r.W2, 0, CCH, t1, 0, CCH, r.b2, BDIM * NPTS, CCH, CCH, 1.0f, 0, 1);
    colstats(t1, 1e-5f);
    apply_cnorm(t1, xio, xio, BN_SC, 1);
  };

  // CSM operates on [B,C,P] rows (sequence = C, d_model = P)
  auto csm_fn = [&](const CSMP& cp, const float* xdin, float* out) {
    const long sP = (long)CCH * PPOOL;
    rowstats(xdin, BDIM * CCH, PPOOL, 1e-3f);
    apply_norm(xdin, nullptr, c_pre, BDIM * CCH, PPOOL, BN_SC, 1);
    gemm_cn(cp.c1W, 0, CCH, c_pre, sP, PPOOL, c_o1, sP, PPOOL, cp.c1b, CCH, PPOOL, CCH, 1.0f, 0, BDIM);
    rowstats(c_o1, BDIM * CCH, PPOOL, 1e-5f);
    apply_ln(c_o1, cp.ln_g, cp.ln_b, c_on, BDIM * CCH, PPOOL);
    run_mamba(cp.sam, c_on, c_mo, BDIM, CCH);
    addk(c_o1, c_mo, c_o1, PSZ);
    rowstats(c_o1, BDIM * CCH, PPOOL, 1e-3f);
    apply_norm(c_o1, nullptr, c_pre, BDIM * CCH, PPOOL, BN_SC, 1);
    gemm_cn(cp.c3W, 0, CCH, c_pre, sP, PPOOL, c_on, sP, PPOOL, cp.c3b, CCH, PPOOL, CCH, 1.0f, 0, BDIM);
    addk(c_on, xdin, out, PSZ);
  };

  auto cssm_fn = [&](const CSSMP& cp) {
    mamba_block(cp.mb_ln_g, cp.mb_ln_b, cp.mbm, cur, x12);
    k_transpose<<<ewg(ACT), 256, 0, stream>>>(cur, xcn, BDIM, NPTS, CCH); // [B,C,N]
    // pre = relu(BN*inorm(x,1e-3)), reused by pool and unpool embeddings
    colstats(cur, 1e-3f);
    apply_cnorm(cur, nullptr, t2, BN_SC, 1);
    // pool embedding [B,N,P] -> transpose -> softmax over n (rows of [B,P,N])
    gemm_rr(t2, 0, CCH, cp.poolW, 0, CCH, emb2, 0, PPOOL, cp.poolb,
            BDIM * NPTS, PPOOL, CCH, 1.0f, 0, 1);
    k_transpose<<<ewg(EMBS), 256, 0, stream>>>(emb2, emb, BDIM, NPTS, PPOOL); // [B,P,N]
    k_softmax_rows<<<BDIM * PPOOL, 256, 0, stream>>>(emb, NPTS);
    // diff_pool: xd[b][c][p] = sum_n x[b][c][n] * S[b][p][n]
    gemm_rr(xcn, (long)CCH * NPTS, NPTS, emb, (long)PPOOL * NPTS, NPTS,
            xd, (long)CCH * PPOOL, PPOOL, nullptr, CCH, PPOOL, NPTS, 1.0f, 0, BDIM);
    csm_fn(cp.csm, xd, x2b);
    // unpool embedding [B,N,P], softmax over p = contiguous row softmax
    gemm_rr(t2, 0, CCH, cp.unpoolW, 0, CCH, emb2, 0, PPOOL, cp.unpoolb,
            BDIM * NPTS, PPOOL, CCH, 1.0f, 0, 1);
    k_softmax_rows<<<BDIM * NPTS, 256, 0, stream>>>(emb2, PPOOL);
    // diff_unpool: xu[b][n][c] = sum_p S2[b][n][p] * x2[b][c][p]
    gemm_rr(emb2, (long)NPTS * PPOOL, PPOOL, x2b, (long)CCH * PPOOL, PPOOL,
            xu, (long)NPTS * CCH, CCH, nullptr, NPTS, CCH, PPOOL, 1.0f, 0, BDIM);
    long totc = (long)BDIM * NPTS * 2 * CCH;
    k_concat_rows<<<ewg(totc), 256, 0, stream>>>(cur, xu, catb, totc, CCH);
    gemm_rr(catb, 0, 2 * CCH, cp.shotW, 0, 2 * CCH, t1, 0, CCH, cp.shotb,
            BDIM * NPTS, CCH, 2 * CCH, 1.0f, 0, 1);
    addk(t1, x12, cur, ACT);
  };

  // ---------------------------- forward pass --------------------------------
  k_conv0<<<ewg(ACT), 256, 0, stream>>>(x_in, conv0W, conv0b, cur, ACT, CCH, NPTS);
  for (int i = 0; i < 4; ++i) resblock(lpfe_a[i], cur);

  k_knn<<<dim3(NPTS, BDIM), 256, 0, stream>>>(cur, knn_idx, CCH, NPTS);
  k_dgcnn<<<dim3(NPTS, BDIM), 256, 0, stream>>>(cur, knn_idx, dg.W1, dg.b1, dg.W2, dg.b2,
                                                a_dg, CCH, NPTS);
  { float* tmp = cur; cur = a_dg; a_dg = tmp; }

  for (int i = 0; i < 4; ++i) resblock(lpfe_b[i], cur);
  for (int i = 0; i < 3; ++i) cssm_fn(cs[i]);
  for (int i = 0; i < 4; ++i) resblock(gcc_a[i], cur);

  // CFBM: forward mamba block, reversed mamba block, residual
  mamba_block(cfbm_ln_g, cfbm_ln_b, cfbm_m, cur, f1);
  k_fliprows<<<ewg(ACT), 256, 0, stream>>>(f1, f2, BDIM, NPTS, CCH);
  mamba_block(cfbm_ln_g, cfbm_ln_b, cfbm_m, f2, f1);
  k_fliprows<<<ewg(ACT), 256, 0, stream>>>(f1, f2, BDIM, NPTS, CCH);
  addk(f2, cur, cur, ACT);

  for (int i = 0; i < 4; ++i) resblock(gcc_b[i], cur);
  resblock(embed1, cur);

  long totO = (long)BDIM * NPTS;
  k_lin0<<<ewg(totO), 256, 0, stream>>>(cur, lin0W, lin0b, (float*)d_out, totO, CCH);
}
